// GCN_27539330301990
// MI455X (gfx1250) — matrix operand order repaired and checked by reading, not compiled, not run on hardware
//
#include <hip/hip_runtime.h>
#include <hip/hip_bf16.h>

#define NNODES 50000   // divisible by 16 -> every GEMM tile is full
#define NEDGES 800000

typedef __attribute__((ext_vector_type(16))) __bf16 v16bf;
typedef __attribute__((ext_vector_type(8)))  __bf16 v8bf;
typedef __attribute__((ext_vector_type(8)))  float  v8f;

// ---------------- degree / norm ----------------
__global__ void deg_init_kernel(float* deg, int n) {
    int i = blockIdx.x * blockDim.x + threadIdx.x;
    if (i < n) deg[i] = 1.0f;   // self loop contributes 1 to every node
}

__global__ void deg_count_kernel(const int* __restrict__ dst, float* deg, int e) {
    int i = blockIdx.x * blockDim.x + threadIdx.x;
    int stride = gridDim.x * blockDim.x;
    for (; i < e; i += stride) atomicAdd(&deg[dst[i]], 1.0f);
}

__global__ void deg_rsqrt_kernel(float* deg, int n) {
    int i = blockIdx.x * blockDim.x + threadIdx.x;
    if (i < n) {
        float d = deg[i];
        deg[i] = (d > 0.0f) ? rsqrtf(d) : 0.0f;
    }
}

// ---------------- dense GEMM H = X @ W via bf16 WMMA ----------------
// One wave computes a 16-row stripe of H across all NCOLS columns.
// K = 128 fixed. Block = 128 threads = 4 waves = 64 rows.
// LDS holds W pre-swizzled into WMMA B-fragment order:
//   sB[nt][kk][lane][i] = bf16( W[(kk*32 + (lane>>4)*16 + i) * NCOLS
//                                 + nt*16 + (lane&15)] )
// so each lane's 16-element fragment is 32 contiguous, 32B-aligned bytes.
// nrows must be a multiple of 16 (true here), so stores are unguarded.
template <int NCOLS>
__global__ void gemm_wmma_kernel(const float* __restrict__ X,
                                 const float* __restrict__ W,
                                 float* __restrict__ H, int nrows) {
    constexpr int NT = NCOLS / 16;
    __shared__ __bf16 sB[NT * 4 * 32 * 16];   // == 128*NCOLS bf16 (32KB max)

    const int tid = threadIdx.x;
    // stage: each thread packs 8 contiguous fragment entries -> one 16B store
    for (int g = tid; g < NT * 4 * 32 * 2; g += blockDim.x) {
        const int i0    = (g & 1) * 8;
        const int slane = (g >> 1) & 31;
        const int kk    = (g >> 6) & 3;
        const int nt    = g >> 8;
        const int kbase = kk * 32 + (slane >> 4) * 16 + i0;
        const int col   = nt * 16 + (slane & 15);
        v8bf p;
#pragma unroll
        for (int i = 0; i < 8; ++i) p[i] = (__bf16)W[(kbase + i) * NCOLS + col];
        *reinterpret_cast<v8bf*>(&sB[g * 8]) = p;
    }
    __syncthreads();

    const int wave = tid >> 5;
    const int lane = tid & 31;
    const int row0 = (blockIdx.x * 4 + wave) * 16;
    if (row0 >= nrows) return;           // wave-uniform exit (after barrier)

    // ---- A fragments: 16x32 bf16, contiguous layout:
    // lanes 0-15: row M=lane, K = 0..15; lanes 16-31: K = 16..31
    const int mrow  = lane & 15;
    const int khalf = lane >> 4;
    const int grow  = row0 + mrow;                // in-bounds: 16 | nrows
    const float* xr = X + (size_t)grow * 128;

    v16bf afrag[4];
#pragma unroll
    for (int kk = 0; kk < 4; ++kk) {
        const int kb0 = kk * 32 + khalf * 16;
        const int kb1 = kb0 + 8;
#pragma unroll
        for (int i = 0; i < 8; ++i) {
            afrag[kk][i]     = (__bf16)xr[kb0 + i];
            afrag[kk][i + 8] = (__bf16)xr[kb1 + i];
        }
    }

    // C layout: VGPR r -> row (rbase + r), col = lane&15
    const int rbase = row0 + khalf * 8;
#pragma unroll
    for (int nt = 0; nt < NT; ++nt) {
        v8f c = {};
#pragma unroll
        for (int kk = 0; kk < 4; ++kk) {
            v16bf bfrag = *reinterpret_cast<const v16bf*>(
                &sB[((nt * 4 + kk) * 32 + lane) * 16]);
            c = __builtin_amdgcn_wmma_f32_16x16x32_bf16(
                    false, afrag[kk], false, bfrag, (short)0, c, false, false);
        }
        float* hp = H + (size_t)rbase * NCOLS + nt * 16 + (lane & 15);
#pragma unroll
        for (int r = 0; r < 8; ++r) hp[r * NCOLS] = c[r];  // imm-offset stores
    }
}

// ---------------- init output rows with bias ----------------
template <int F>
__global__ void bias_init_kernel(float* __restrict__ out,
                                 const float* __restrict__ b, int nrows) {
    size_t i = (size_t)blockIdx.x * blockDim.x + threadIdx.x;
    size_t total = (size_t)nrows * F;
    size_t stride = (size_t)gridDim.x * blockDim.x;
    for (; i < total; i += stride) out[i] = b[i & (F - 1)];
}

// ---------------- edge aggregation: out[dst] += h[src] * norm ----------------
// One wave per edge; lane handles F/32 contiguous floats.
template <int F>
__global__ void aggregate_kernel(const float* __restrict__ H,
                                 const int* __restrict__ src,
                                 const int* __restrict__ dst,
                                 const float* __restrict__ dis,
                                 float* __restrict__ out,
                                 int nedges, int ntotal) {
    const int lane = threadIdx.x & 31;
    int wid = blockIdx.x * (blockDim.x >> 5) + (threadIdx.x >> 5);
    const int nw = gridDim.x * (blockDim.x >> 5);
    constexpr int PER = F / 32;
    for (int e = wid; e < ntotal; e += nw) {
        int s, d;
        if (e < nedges) { s = src[e]; d = dst[e]; }
        else            { s = e - nedges; d = s; }        // self loop
        float w = dis[s] * dis[d];
        const float* hp = H + (size_t)s * F + lane * PER;
        float* op = out + (size_t)d * F + lane * PER;
        if constexpr (PER == 4) {
            float4 v = *reinterpret_cast<const float4*>(hp);
            atomicAdd(op + 0, v.x * w);
            atomicAdd(op + 1, v.y * w);
            atomicAdd(op + 2, v.z * w);
            atomicAdd(op + 3, v.w * w);
        } else {
            float2 v = *reinterpret_cast<const float2*>(hp);
            atomicAdd(op + 0, v.x * w);
            atomicAdd(op + 1, v.y * w);
        }
    }
}

// ---------------- batchnorm ----------------
__global__ void zero_stats_kernel(float* s) {
    if (threadIdx.x < 256) s[threadIdx.x] = 0.0f;   // s1[128] ++ s2[128]
}

__global__ void bn_stats_kernel(const float* __restrict__ x,
                                float* __restrict__ s1,
                                float* __restrict__ s2, int nrows) {
    const int col = threadIdx.x & 127;
    int r = blockIdx.x * 2 + (threadIdx.x >> 7);
    const int rstride = gridDim.x * 2;
    float a = 0.0f, b = 0.0f;
    for (; r < nrows; r += rstride) {
        float v = x[(size_t)r * 128 + col];
        a += v;
        b += v * v;
    }
    atomicAdd(&s1[col], a);
    atomicAdd(&s2[col], b);
}

// rewrite (sum, sumsq) in place as (scale, shift): y = x*scale + shift
__global__ void bn_finalize_kernel(float* s1, float* s2,
                                   const float* __restrict__ gamma,
                                   const float* __restrict__ beta) {
    int col = threadIdx.x;
    if (col < 128) {
        const float invN = 1.0f / (float)NNODES;
        float mean = s1[col] * invN;
        float var  = s2[col] * invN - mean * mean;
        float sc   = gamma[col] * rsqrtf(var + 1e-5f);
        s1[col] = sc;
        s2[col] = beta[col] - mean * sc;
    }
}

__global__ void bn_relu_kernel(const float* __restrict__ x,
                               const float* __restrict__ sc,
                               const float* __restrict__ sh,
                               float* __restrict__ y, int nrows) {
    size_t i = (size_t)blockIdx.x * blockDim.x + threadIdx.x;
    size_t total = (size_t)nrows * 128;
    size_t stride = (size_t)gridDim.x * blockDim.x;
    for (; i < total; i += stride) {
        int col = (int)(i & 127);
        float v = fmaf(x[i], sc[col], sh[col]);
        y[i] = v > 0.0f ? v : 0.0f;
    }
}

extern "C" void kernel_launch(void* const* d_in, const int* in_sizes, int n_in,
                              void* d_out, int out_size, void* d_ws, size_t ws_size,
                              hipStream_t stream) {
    const float* x   = (const float*)d_in[0];
    const int*   ei  = (const int*)d_in[1];
    const int*   src = ei;
    const int*   dst = ei + NEDGES;
    const float* W1  = (const float*)d_in[2];
    const float* b1  = (const float*)d_in[3];
    const float* g1  = (const float*)d_in[4];
    const float* be1 = (const float*)d_in[5];
    const float* W2  = (const float*)d_in[6];
    const float* b2  = (const float*)d_in[7];
    const float* g2  = (const float*)d_in[8];
    const float* be2 = (const float*)d_in[9];
    const float* W3  = (const float*)d_in[10];
    const float* b3  = (const float*)d_in[11];
    float* out = (float*)d_out;

    float* ws   = (float*)d_ws;
    float* buf0 = ws;                               // N*128
    float* buf1 = buf0 + (size_t)NNODES * 128;      // N*128
    float* dis  = buf1 + (size_t)NNODES * 128;      // N (deg -> d_isqrt in place)
    float* s1   = dis + NNODES;                     // 128
    float* s2   = s1 + 128;                         // 128

    const int ntotal = NEDGES + NNODES;
    const int gemm_blocks = ((NNODES + 15) / 16 + 3) / 4;
    const int agg_blocks  = (ntotal + 7) / 8;       // 8 waves per 256-thr block
    const int el_blocks128 = (int)(((size_t)NNODES * 128 + 255) / 256);
    const int el_blocks64  = (int)(((size_t)NNODES * 64 + 255) / 256);

    // ---- normalization coefficients
    deg_init_kernel<<<(NNODES + 255) / 256, 256, 0, stream>>>(dis, NNODES);
    deg_count_kernel<<<(NEDGES + 255) / 256, 256, 0, stream>>>(dst, dis, NEDGES);
    deg_rsqrt_kernel<<<(NNODES + 255) / 256, 256, 0, stream>>>(dis, NNODES);

    // ---- layer 1: conv -> BN -> ReLU
    gemm_wmma_kernel<128><<<gemm_blocks, 128, 0, stream>>>(x, W1, buf0, NNODES);
    bias_init_kernel<128><<<el_blocks128, 256, 0, stream>>>(buf1, b1, NNODES);
    aggregate_kernel<128><<<agg_blocks, 256, 0, stream>>>(buf0, src, dst, dis,
                                                          buf1, NEDGES, ntotal);
    zero_stats_kernel<<<1, 256, 0, stream>>>(s1);
    bn_stats_kernel<<<512, 256, 0, stream>>>(buf1, s1, s2, NNODES);
    bn_finalize_kernel<<<1, 128, 0, stream>>>(s1, s2, g1, be1);
    bn_relu_kernel<<<el_blocks128, 256, 0, stream>>>(buf1, s1, s2, buf0, NNODES);

    // ---- layer 2: conv -> BN -> ReLU
    gemm_wmma_kernel<128><<<gemm_blocks, 128, 0, stream>>>(buf0, W2, buf1, NNODES);
    bias_init_kernel<128><<<el_blocks128, 256, 0, stream>>>(buf0, b2, NNODES);
    aggregate_kernel<128><<<agg_blocks, 256, 0, stream>>>(buf1, src, dst, dis,
                                                          buf0, NEDGES, ntotal);
    zero_stats_kernel<<<1, 256, 0, stream>>>(s1);
    bn_stats_kernel<<<512, 256, 0, stream>>>(buf0, s1, s2, NNODES);
    bn_finalize_kernel<<<1, 128, 0, stream>>>(s1, s2, g2, be2);
    bn_relu_kernel<<<el_blocks128, 256, 0, stream>>>(buf0, s1, s2, buf1, NNODES);

    // ---- layer 3: conv only, aggregate straight into d_out
    gemm_wmma_kernel<64><<<gemm_blocks, 128, 0, stream>>>(buf1, W3, buf0, NNODES);
    bias_init_kernel<64><<<el_blocks64, 256, 0, stream>>>(out, b3, NNODES);
    aggregate_kernel<64><<<agg_blocks, 256, 0, stream>>>(buf0, src, dst, dis,
                                                         out, NEDGES, ntotal);
}
